// ReAttention_37426345017463
// MI455X (gfx1250) — matrix operand order repaired and checked by reading, not compiled, not run on hardware
//
#include <hip/hip_runtime.h>
#include <hip/hip_bf16.h>

// ---------------------------------------------------------------------------
// ReAttention for MI455X (gfx1250, wave32, WMMA bf16 16x16x32, fp32 accum)
// B=4 N=1024 D=768 H=12 Dh=64
//
// CDNA5 data paths used:
//   - v_wmma_f32_16x16x32_bf16 for all GEMMs (fp32 accumulate)
//   - global_load_async_to_lds_b128 (ASYNCcnt) for A-tile staging in K1
//   - tensor_load_to_lds (TDM, TENSORcnt, 6-arg form) for V-panel staging
//     in K4, double-buffered across the K-loop
//   - global_prefetch for next W tile in K1
//
// ws layout (bf16): xb, wqb/wkb/wvb, qb/kb, vt (transposed), S 96MB, AT 96MB.
// The 2x96MB attention intermediates fit in the 192MB L2.
// ---------------------------------------------------------------------------

typedef __attribute__((ext_vector_type(16))) __bf16 v16bf;
typedef __attribute__((ext_vector_type(8)))  float  v8f;
typedef __attribute__((ext_vector_type(4)))  unsigned int v4u;
typedef __attribute__((ext_vector_type(8)))  int v8i;
typedef __attribute__((ext_vector_type(4)))  int v4i;

#define B_   4
#define N_   1024
#define D_   768
#define H_   12
#define DH_  64
#define HD_  768   /* H_*DH_ */
#define M_   4096  /* B_*N_  */

// typed, address-space-qualified pointer casts for the async-LDS builtins
#define GAS1_V4I(p) ((__attribute__((address_space(1))) v4i*)(void*)(p))
#define LAS3_V4I(p) ((__attribute__((address_space(3))) v4i*)(void*)(p))
#define LDS_OFF(p) ((unsigned)(unsigned long long)(__attribute__((address_space(3))) void*)(void*)(p))

__device__ __forceinline__ __bf16 f2bf(float f) {
  unsigned u = __builtin_bit_cast(unsigned, f);
  unsigned r = u + 0x7FFFu + ((u >> 16) & 1u);     // round-to-nearest-even
  unsigned short s = (unsigned short)(r >> 16);
  return __builtin_bit_cast(__bf16, s);
}
__device__ __forceinline__ float bf2f(__bf16 b) {
  unsigned short s = __builtin_bit_cast(unsigned short, b);
  unsigned u = ((unsigned)s) << 16;
  return __builtin_bit_cast(float, u);
}

union Frag { v16bf v; uint4 q[2]; unsigned short s[16]; };

// ---------------------------------------------------------------------------
// TDM: 2D tile load  [tile_d rows x tile_q cols] of bf16 from a row-major
// tensor with row stride `stride_q` elements, into LDS at lds_off.
// Descriptor layout per CDNA5 ISA ch.8 (groups 0/1; trailing groups zero
// for a 2D tensor). This toolchain's builtin takes the 6-arg form.
// ---------------------------------------------------------------------------
__device__ __forceinline__ void tdm_load_tile_bf16(const __bf16* gsrc,
                                                   unsigned lds_off,
                                                   unsigned tile_q, unsigned tile_d,
                                                   unsigned stride_q) {
  unsigned long long ga = (unsigned long long)(void*)gsrc;
  v4u g0;
  g0[0] = 1u;                                   // count=1, user descriptor
  g0[1] = lds_off;                              // lds_addr (bytes)
  g0[2] = (unsigned)ga;                         // global_addr[31:0]
  g0[3] = (unsigned)(ga >> 32) | (2u << 30);    // global_addr[56:32] | type=2
  v8i g1;
  g1[0] = 0x00010000;                           // wg_mask=0, data_size=1 (2B)
  g1[1] = (int)((stride_q & 0xFFFFu) << 16);    // tensor_dim0[15:0] @bits63:48
  g1[2] = (int)(((stride_q >> 16) & 0xFFFFu)    // tensor_dim0[31:16]
           | ((tile_d & 0xFFFFu) << 16));       // tensor_dim1[15:0] @bits95:80
  g1[3] = (int)((tile_q & 0xFFFFu) << 16);      // tile_dim0 @bits127:112
  g1[4] = (int)(tile_d & 0xFFFFu);              // tile_dim1 @bits143:128
  g1[5] = (int)stride_q;                        // tensor_dim0_stride[31:0]
  g1[6] = 0;                                    // stride hi / dim1_stride lo
  g1[7] = 0;
  v4i z4 = {0, 0, 0, 0};
  v8i z8 = {0, 0, 0, 0, 0, 0, 0, 0};
  __builtin_amdgcn_tensor_load_to_lds(g0, g1, z4, z4, z8, 0);
}

// ---------------------------------------------------------------------------
// K0: fp32 -> bf16 pack (optionally scaled)
// ---------------------------------------------------------------------------
__global__ __launch_bounds__(256) void pack_bf16(const float* __restrict__ src,
                                                 __bf16* __restrict__ dst,
                                                 int n, float scale) {
  int i = blockIdx.x * 256 + threadIdx.x;
  if (i < n) dst[i] = f2bf(src[i] * scale);
}

// ---------------------------------------------------------------------------
// K1: QKV projection GEMM  C[4096,768] = X[4096,768] * W[768,768]
// block = 256 thr (8 waves), tile 128(M) x 64(N), K-steps of 32.
// A tile staged with async global->LDS copies (ASYNCcnt); B tile staged
// transposed via regular loads + ds stores; next W tile prefetched.
// ---------------------------------------------------------------------------
__global__ __launch_bounds__(256) void gemm_qkv(const __bf16* __restrict__ X,
                                                const __bf16* __restrict__ W,
                                                __bf16* __restrict__ out,
                                                int vt_mode) {
  __shared__ __bf16 As[128][32];   // [m][k]
  __shared__ __bf16 Bs[64][32];    // [n][k]  (transposed stage)
  const int t    = threadIdx.x;
  const int wave = t >> 5, lane = t & 31;
  const int lm   = lane & 15, lh = lane >> 4;
  const int m0   = blockIdx.y * 128;
  const int n0   = blockIdx.x * 64;

  v8f acc[4] = {};

  for (int kt = 0; kt < D_; kt += 32) {
    __syncthreads();
    { // stage A: 128x32 bf16 via async copy, each thread 32B (2 x b128)
      int row = t >> 1, kc = (t & 1) * 16;
      const __bf16* g = X + (size_t)(m0 + row) * D_ + kt + kc;
      __bf16* l = &As[row][kc];
      __builtin_amdgcn_global_load_async_to_lds_b128(GAS1_V4I(g), LAS3_V4I(l), 0, 0);
      __builtin_amdgcn_global_load_async_to_lds_b128(GAS1_V4I(g), LAS3_V4I(l), 16, 0);
    }
    { // stage B transposed: W[k][n] -> Bs[n][k]; prefetch next W tile
      int e0 = t * 8, kk = e0 >> 6, nn = e0 & 63;
      union { uint4 u; unsigned short s[8]; } tmp;
      tmp.u = *(const uint4*)(W + (size_t)(kt + kk) * HD_ + n0 + nn);
      #pragma unroll
      for (int j = 0; j < 8; ++j)
        Bs[nn + j][kk] = __builtin_bit_cast(__bf16, tmp.s[j]);
      if (kt + 32 < D_)
        __builtin_prefetch(W + (size_t)(kt + 32 + kk) * HD_ + n0 + nn, 0, 3);
    }
    asm volatile("s_wait_asynccnt 0x0" ::: "memory");
    __syncthreads();

    Frag a;
    { // A fragment: lane<16 -> K 0..15 ; lane>=16 -> K 16..31
      int row = wave * 16 + lm;
      int k0  = lh * 16;
      a.q[0] = *(const uint4*)(&As[row][k0]);
      a.q[1] = *(const uint4*)(&As[row][k0 + 8]);
    }
    #pragma unroll
    for (int g = 0; g < 4; ++g) {
      Frag b; // B fragment: lane<16 -> K 0..15 ; lane>=16 -> K 16..31
      int col = g * 16 + lm;
      int k0  = lh * 16;
      b.q[0] = *(const uint4*)(&Bs[col][k0]);
      b.q[1] = *(const uint4*)(&Bs[col][k0 + 8]);
      acc[g] = __builtin_amdgcn_wmma_f32_16x16x32_bf16(
          false, a.v, false, b.v, (short)0, acc[g], false, false);
    }
  }

  #pragma unroll
  for (int g = 0; g < 4; ++g) {
    int col = n0 + g * 16 + lm;
    #pragma unroll
    for (int r = 0; r < 8; ++r) {
      int row = m0 + wave * 16 + (lh ? 8 + r : r);
      __bf16 v = f2bf(acc[g][r]);
      if (!vt_mode) {
        out[(size_t)row * HD_ + col] = v;
      } else {
        int b  = row >> 10, nq = row & 1023;
        int h  = col >> 6,  d  = col & 63;
        out[(size_t)((b * H_ + h) * DH_ + d) * N_ + nq] = v;
      }
    }
  }
}

// ---------------------------------------------------------------------------
// K2: attention scores  S[b,h,q,k] = sum_d q[b,q,h,d] * k[b,k,h,d]
// block 256 thr: tile 128(q) x 128(k) per (b,h). wave w -> 16 q-rows x 128 k.
// K = Dh = 64 -> 2 WMMA k-steps, fragments direct from global (L2-resident).
// ---------------------------------------------------------------------------
__global__ __launch_bounds__(256) void attn_scores(const __bf16* __restrict__ Q,
                                                   const __bf16* __restrict__ K,
                                                   __bf16* __restrict__ S) {
  const int t = threadIdx.x, wave = t >> 5, lane = t & 31;
  const int lm = lane & 15, lh = lane >> 4;
  const int bh = blockIdx.z, b = bh / H_, h = bh % H_;
  const int q0 = blockIdx.y * 128 + wave * 16;
  const int k0 = blockIdx.x * 128;

  v8f acc[8] = {};
  #pragma unroll
  for (int step = 0; step < 2; ++step) {
    Frag a;
    {
      int qrow = q0 + lm;
      int d0   = step * 32 + lh * 16;
      const __bf16* p = Q + (size_t)((b * N_ + qrow) * H_ + h) * DH_ + d0;
      a.q[0] = *(const uint4*)p;
      a.q[1] = *(const uint4*)(p + 8);
    }
    #pragma unroll
    for (int g = 0; g < 8; ++g) {
      Frag bf;
      int key = k0 + g * 16 + lm;
      int d0  = step * 32 + lh * 16;
      const __bf16* p = K + (size_t)((b * N_ + key) * H_ + h) * DH_ + d0;
      bf.q[0] = *(const uint4*)p;
      bf.q[1] = *(const uint4*)(p + 8);
      acc[g] = __builtin_amdgcn_wmma_f32_16x16x32_bf16(
          false, a.v, false, bf.v, (short)0, acc[g], false, false);
    }
  }
  #pragma unroll
  for (int g = 0; g < 8; ++g) {
    int key = k0 + g * 16 + lm;
    #pragma unroll
    for (int r = 0; r < 8; ++r) {
      int qrow = q0 + (lh ? 8 + r : r);
      S[((size_t)(bh * N_ + qrow)) * N_ + key] = f2bf(acc[g][r]);
    }
  }
}

// ---------------------------------------------------------------------------
// K3: fused softmax (over k) + theta head-mix + LayerNorm (over k).
// One block per (b,q): 12x1024 scores in 48KB LDS. Writes AT transposed
// [b,i,k,q] so K4's A-fragments are contiguous per lane.
// ---------------------------------------------------------------------------
__global__ __launch_bounds__(256) void softmax_theta_ln(
    const __bf16* __restrict__ S, const float* __restrict__ theta,
    const float* __restrict__ lns, const float* __restrict__ lnb,
    __bf16* __restrict__ AT) {
  __shared__ float P[H_][N_];    // 48 KB
  __shared__ float red[256];
  __shared__ float th[H_ * H_];
  const int t = threadIdx.x;
  const int q = blockIdx.x, b = blockIdx.y;

  if (t < H_ * H_) th[t] = theta[t];

  // --- per-head softmax over k ---
  for (int h = 0; h < H_; ++h) {
    const __bf16* row = S + ((size_t)((b * H_ + h) * N_ + q)) * N_;
    float vals[4];
    float lmax = -3.0e38f;
    #pragma unroll
    for (int j = 0; j < 4; ++j) {
      float v = bf2f(row[t + j * 256]);
      vals[j] = v;
      lmax = fmaxf(lmax, v);
    }
    red[t] = lmax; __syncthreads();
    for (int s2 = 128; s2 > 0; s2 >>= 1) {
      if (t < s2) red[t] = fmaxf(red[t], red[t + s2]);
      __syncthreads();
    }
    float m = red[0]; __syncthreads();
    float lsum = 0.f;
    #pragma unroll
    for (int j = 0; j < 4; ++j) { float e = __expf(vals[j] - m); vals[j] = e; lsum += e; }
    red[t] = lsum; __syncthreads();
    for (int s2 = 128; s2 > 0; s2 >>= 1) {
      if (t < s2) red[t] += red[t + s2];
      __syncthreads();
    }
    float inv = 1.0f / red[0]; __syncthreads();
    #pragma unroll
    for (int j = 0; j < 4; ++j) P[h][t + j * 256] = vals[j] * inv;
  }
  __syncthreads();

  // --- theta mix: A[i,k] = sum_h theta[h,i] * P[h,k]  (in place per column) ---
  for (int j = 0; j < 4; ++j) {
    int kk = t + j * 256;
    float p[H_], o[H_];
    #pragma unroll
    for (int h = 0; h < H_; ++h) p[h] = P[h][kk];
    #pragma unroll
    for (int i = 0; i < H_; ++i) {
      float a = 0.f;
      #pragma unroll
      for (int h = 0; h < H_; ++h) a += th[h * H_ + i] * p[h];
      o[i] = a;
    }
    #pragma unroll
    for (int i = 0; i < H_; ++i) P[i][kk] = o[i];
  }
  __syncthreads();

  // --- LayerNorm over k per output head i, then store transposed bf16 ---
  for (int i = 0; i < H_; ++i) {
    float ls = 0.f;
    #pragma unroll
    for (int j = 0; j < 4; ++j) ls += P[i][t + j * 256];
    red[t] = ls; __syncthreads();
    for (int s2 = 128; s2 > 0; s2 >>= 1) {
      if (t < s2) red[t] += red[t + s2];
      __syncthreads();
    }
    float mean = red[0] * (1.0f / N_); __syncthreads();
    float lv = 0.f;
    #pragma unroll
    for (int j = 0; j < 4; ++j) { float d = P[i][t + j * 256] - mean; lv += d * d; }
    red[t] = lv; __syncthreads();
    for (int s2 = 128; s2 > 0; s2 >>= 1) {
      if (t < s2) red[t] += red[t + s2];
      __syncthreads();
    }
    float rstd = rsqrtf(red[0] * (1.0f / N_) + 1e-6f); __syncthreads();
    #pragma unroll
    for (int j = 0; j < 4; ++j) {
      int kk = t + j * 256;
      float v = (P[i][kk] - mean) * rstd * lns[kk] + lnb[kk];
      AT[((size_t)(b * H_ + i) * N_ + kk) * N_ + q] = f2bf(v);
    }
  }
}

// ---------------------------------------------------------------------------
// K4: out[b,k,h,d] = sum_q AT[b,h,k,q] * v[b,q,h,d]
// A = AT row-major [k,q]; B panel [64 d x 32 q] staged per K-step by the
// Tensor Data Mover into double-buffered LDS (wave 0 issues the descriptor,
// waits TENSORcnt, block barrier hands off). 128 k-rows x 64 d per block.
// ---------------------------------------------------------------------------
__global__ __launch_bounds__(256) void attn_out(const __bf16* __restrict__ AT,
                                                const __bf16* __restrict__ VT,
                                                float* __restrict__ out) {
  __shared__ __bf16 Bbuf[2][DH_ * 32];   // 2 x 4KB: [d][q(32)]
  const int t = threadIdx.x, wave = t >> 5, lane = t & 31;
  const int lm = lane & 15, lh = lane >> 4;
  const int bh = blockIdx.y, b = bh / H_, h = bh % H_;
  const int kp0 = blockIdx.x * 128 + wave * 16;
  const __bf16* vbase = VT + (size_t)bh * DH_ * N_;

  // prologue: TDM V-panel for q0=0 into buffer 0
  if (wave == 0) {
    tdm_load_tile_bf16(vbase, LDS_OFF(&Bbuf[0][0]), 32, DH_, N_);
    __builtin_amdgcn_s_wait_tensorcnt(0);
  }
  __syncthreads();

  v8f acc[4] = {};
  for (int q0 = 0; q0 < N_; q0 += 32) {
    const int cur = (q0 >> 5) & 1;
    if (wave == 0 && (q0 + 32) < N_)   // TDM prefetch next panel
      tdm_load_tile_bf16(vbase + q0 + 32, LDS_OFF(&Bbuf[cur ^ 1][0]), 32, DH_, N_);

    Frag a;
    {
      int krow = kp0 + lm;
      const __bf16* p = AT + ((size_t)(bh * N_ + krow)) * N_ + q0 + lh * 16;
      a.q[0] = *(const uint4*)p;
      a.q[1] = *(const uint4*)(p + 8);
    }
    #pragma unroll
    for (int g = 0; g < 4; ++g) {
      Frag bf;
      int d = g * 16 + lm;
      const __bf16* p = &Bbuf[cur][d * 32 + lh * 16];
      bf.q[0] = *(const uint4*)p;
      bf.q[1] = *(const uint4*)(p + 8);
      acc[g] = __builtin_amdgcn_wmma_f32_16x16x32_bf16(
          false, a.v, false, bf.v, (short)0, acc[g], false, false);
    }

    if (wave == 0) __builtin_amdgcn_s_wait_tensorcnt(0);
    __syncthreads();
  }

  #pragma unroll
  for (int g = 0; g < 4; ++g) {
    int d = g * 16 + lm;
    #pragma unroll
    for (int r = 0; r < 8; ++r) {
      int kp = kp0 + (lh ? 8 + r : r);
      out[(size_t)((b * N_ + kp) * H_ + h) * DH_ + d] = acc[g][r];
    }
  }
}

// ---------------------------------------------------------------------------
extern "C" void kernel_launch(void* const* d_in, const int* in_sizes, int n_in,
                              void* d_out, int out_size, void* d_ws, size_t ws_size,
                              hipStream_t stream) {
  (void)in_sizes; (void)n_in; (void)out_size; (void)ws_size;
  const float* x     = (const float*)d_in[0];
  const float* Wq    = (const float*)d_in[1];
  const float* Wk    = (const float*)d_in[2];
  const float* Wv    = (const float*)d_in[3];
  const float* theta = (const float*)d_in[4];
  const float* lns   = (const float*)d_in[5];
  const float* lnb   = (const float*)d_in[6];
  float* out = (float*)d_out;

  char* ws = (char*)d_ws;
  size_t off = 0;
  auto alloc = [&](size_t bytes) -> void* {
    void* p = ws + off;
    off += (bytes + 255) & ~(size_t)255;
    return p;
  };
  __bf16* xb  = (__bf16*)alloc((size_t)M_ * D_ * 2);
  __bf16* wqb = (__bf16*)alloc((size_t)D_ * HD_ * 2);
  __bf16* wkb = (__bf16*)alloc((size_t)D_ * HD_ * 2);
  __bf16* wvb = (__bf16*)alloc((size_t)D_ * HD_ * 2);
  __bf16* qb  = (__bf16*)alloc((size_t)M_ * HD_ * 2);
  __bf16* kb  = (__bf16*)alloc((size_t)M_ * HD_ * 2);
  __bf16* vt  = (__bf16*)alloc((size_t)M_ * HD_ * 2);
  __bf16* sb  = (__bf16*)alloc((size_t)B_ * H_ * N_ * N_ * 2);   // 96 MB
  __bf16* at  = (__bf16*)alloc((size_t)B_ * H_ * N_ * N_ * 2);   // 96 MB

  // K0: packs (fold 1/sqrt(Dh)=1/8 into Wq)
  pack_bf16<<<(M_ * D_ + 255) / 256, 256, 0, stream>>>(x, xb, M_ * D_, 1.0f);
  pack_bf16<<<(D_ * HD_ + 255) / 256, 256, 0, stream>>>(Wq, wqb, D_ * HD_, 0.125f);
  pack_bf16<<<(D_ * HD_ + 255) / 256, 256, 0, stream>>>(Wk, wkb, D_ * HD_, 1.0f);
  pack_bf16<<<(D_ * HD_ + 255) / 256, 256, 0, stream>>>(Wv, wvb, D_ * HD_, 1.0f);

  // K1: QKV projections
  dim3 gq(HD_ / 64, M_ / 128);
  gemm_qkv<<<gq, 256, 0, stream>>>(xb, wqb, qb, 0);
  gemm_qkv<<<gq, 256, 0, stream>>>(xb, wkb, kb, 0);
  gemm_qkv<<<gq, 256, 0, stream>>>(xb, wvb, vt, 1);

  // K2: scores
  attn_scores<<<dim3(N_ / 128, N_ / 128, B_ * H_), 256, 0, stream>>>(qb, kb, sb);

  // K3: softmax + theta mix + LayerNorm (fused per (b,q))
  softmax_theta_ln<<<dim3(N_, B_), 256, 0, stream>>>(sb, theta, lns, lnb, at);

  // K4: output GEMM with TDM-staged V panels
  attn_out<<<dim3(N_ / 128, B_ * H_), 256, 0, stream>>>(at, vt, out);
}
